// EfficientProbing_39582418600161
// MI455X (gfx1250) — matrix operand order repaired and checked by reading, not compiled, not run on hardware
//
#include <hip/hip_runtime.h>
#include <hip/hip_bf16.h>

#define DIM   768
#define NH    12
#define HD    64
#define LAT   32
#define NSEQ  4096
#define NB    32
#define CHUNK 24
#define SCALE 0.125f

// padded LDS pitches (break 64-bank conflicts)
#define XP 776   // x tile row pitch (bf16)
#define QP 776   // q tile row pitch (bf16)
#define PP 132   // p tile row pitch (f32)

typedef __attribute__((ext_vector_type(16))) __bf16       v16bf;
typedef __attribute__((ext_vector_type(8)))  float        v8f;
typedef __attribute__((ext_vector_type(4)))  unsigned int uint4v;

union V16 { v16bf v; uint4v q[2]; };

__device__ __forceinline__ v8f wmma_bf16(v16bf a, v16bf b, v8f c) {
  // D(f32 16x16) = A(bf16 16x32) * B(bf16 32x16) + C
  return __builtin_amdgcn_wmma_f32_16x16x32_bf16(
      /*neg_a=*/false, a, /*neg_b=*/false, b,
      /*c_mod=*/(short)0, c, /*reuse_a=*/false, /*reuse_b=*/false);
}

// ---------------------------------------------------------------------------
// K1: per (b,h): S = scale * q_h * k_h^T via WMMA; online softmax stats (m,s)
// per latent row l, merged across lanes/waves in LDS. Reads x once.
// ---------------------------------------------------------------------------
__global__ __launch_bounds__(256) void k1_attn_stats(
    const float* __restrict__ x, const float* __restrict__ latent,
    float2* __restrict__ stats /* [NB][NH][LAT] */) {
  const int b    = blockIdx.x;
  const int h    = blockIdx.y;
  const int wave = threadIdx.x >> 5;
  const int lane = threadIdx.x & 31;
  const int mrow = lane & 15;
  const int khalf = lane >> 4;
  const int kb   = khalf << 3;

  // A fragments: scale*q[l, h*64+k]; 2 l-tiles x 2 k-chunks (K=32 each)
  v16bf a[2][2];
#pragma unroll
  for (int lt = 0; lt < 2; ++lt) {
    const float* q = latent + (size_t)(lt * 16 + mrow) * DIM + h * HD;
#pragma unroll
    for (int kc = 0; kc < 2; ++kc) {
#pragma unroll
      for (int i = 0; i < 16; ++i) {
        int k = kc * 32 + khalf * 16 + i;
        a[lt][kc][i] = (__bf16)(SCALE * q[k]);
      }
    }
  }

  // per-lane online softmax stats over this lane's n-columns
  float mrun[2][8], srun[2][8];
#pragma unroll
  for (int lt = 0; lt < 2; ++lt)
#pragma unroll
    for (int r = 0; r < 8; ++r) { mrun[lt][r] = -3.0e38f; srun[lt][r] = 0.0f; }

  const float* xb = x + (size_t)b * NSEQ * DIM + (size_t)h * HD;

  for (int n0 = wave * 16; n0 < NSEQ; n0 += 128) {
    const float* xrow = xb + (size_t)(n0 + mrow) * DIM;
    __builtin_prefetch(xrow + (size_t)128 * DIM, 0, 0);  // global_prefetch_b8
    // B fragments: k-tile (64 x 16n), lane = column n
    v16bf bk[2];
#pragma unroll
    for (int kc = 0; kc < 2; ++kc) {
      const float* src = xrow + kc * 32 + khalf * 16;
#pragma unroll
      for (int i = 0; i < 16; i += 4) {
        float4 v = *(const float4*)(src + i);
        bk[kc][i]     = (__bf16)v.x; bk[kc][i + 1] = (__bf16)v.y;
        bk[kc][i + 2] = (__bf16)v.z; bk[kc][i + 3] = (__bf16)v.w;
      }
    }
#pragma unroll
    for (int lt = 0; lt < 2; ++lt) {
      v8f c = {};
      c = wmma_bf16(a[lt][0], bk[0], c);
      c = wmma_bf16(a[lt][1], bk[1], c);
#pragma unroll
      for (int r = 0; r < 8; ++r) {        // C layout: vgpr r -> row r (+8 hi half)
        float v = c[r];
        float mnew = fmaxf(mrun[lt][r], v);
        srun[lt][r] = srun[lt][r] * __expf(mrun[lt][r] - mnew) + __expf(v - mnew);
        mrun[lt][r] = mnew;
      }
    }
  }

  // merge 8 waves x 16 lanes per row
  __shared__ float redm[8][32][16];
  __shared__ float reds[8][32][16];
#pragma unroll
  for (int lt = 0; lt < 2; ++lt)
#pragma unroll
    for (int r = 0; r < 8; ++r) {
      redm[wave][lane][lt * 8 + r] = mrun[lt][r];
      reds[wave][lane][lt * 8 + r] = srun[lt][r];
    }
  __syncthreads();
  if (threadIdx.x < LAT) {
    int l = threadIdx.x;
    int lt = l >> 4, row = l & 15;
    int lane0 = (row < 8) ? 0 : 16;       // which half-wave tracked this row
    int r = (row & 7) + lt * 8;
    float M = -3.0e38f;
    for (int w = 0; w < 8; ++w)
      for (int ln = 0; ln < 16; ++ln)
        M = fmaxf(M, redm[w][lane0 + ln][r]);
    float S = 0.0f;
    for (int w = 0; w < 8; ++w)
      for (int ln = 0; ln < 16; ++ln)
        S += reds[w][lane0 + ln][r] * __expf(redm[w][lane0 + ln][r] - M);
    stats[((size_t)b * NH + h) * LAT + l] = make_float2(M, S);
  }
}

// ---------------------------------------------------------------------------
// K2: per (b, 128-row n-tile): stage x-tile to LDS (bf16, once), recompute
// logits for all heads, p = mean_h exp(S-m)/s, then y += p @ x_tile (WMMA),
// accumulated with f32 global atomics. Reads x once (serves both GEMMs).
// LDS: 194KB x + 48.5KB q + 16.5KB p + 3KB stats = 262KB (<320KB/WGP).
// ---------------------------------------------------------------------------
__global__ __launch_bounds__(256) void k2_probs_y(
    const float* __restrict__ x, const float* __restrict__ latent,
    const float2* __restrict__ stats, float* __restrict__ y /* [NB][LAT][DIM] */) {
  extern __shared__ char smem[];
  __bf16* xs = (__bf16*)smem;                                            // [128][XP]
  __bf16* qs = (__bf16*)(smem + (size_t)128 * XP * 2);                   // [LAT][QP]
  float*  ps = (float*)(smem + (size_t)128 * XP * 2 + (size_t)LAT * QP * 2); // [LAT][PP]
  float2* st = (float2*)((char*)ps + (size_t)LAT * PP * 4);              // [NH*LAT]

  const int b    = blockIdx.x;
  const int n0   = blockIdx.y * 128;
  const int tid  = threadIdx.x;
  const int wave = tid >> 5, lane = tid & 31;
  const int mrow = lane & 15, khalf = lane >> 4, kb = khalf << 3;
  const int addl = khalf * 8;

  for (int i = tid; i < NH * LAT; i += 256)
    st[i] = stats[(size_t)b * NH * LAT + i];

  for (int idx = tid; idx < LAT * (DIM / 4); idx += 256) {
    int row = idx / (DIM / 4), c4 = (idx % (DIM / 4)) * 4;
    float4 v = *(const float4*)(latent + (size_t)row * DIM + c4);
    __bf16* d = qs + (size_t)row * QP + c4;
    d[0] = (__bf16)(SCALE * v.x); d[1] = (__bf16)(SCALE * v.y);
    d[2] = (__bf16)(SCALE * v.z); d[3] = (__bf16)(SCALE * v.w);
  }
  const float* xb = x + ((size_t)b * NSEQ + n0) * DIM;
  for (int idx = tid; idx < 128 * (DIM / 4); idx += 256) {
    int row = idx / (DIM / 4), c4 = (idx % (DIM / 4)) * 4;
    float4 v = *(const float4*)(xb + (size_t)row * DIM + c4);
    __bf16* d = xs + (size_t)row * XP + c4;
    d[0] = (__bf16)v.x; d[1] = (__bf16)v.y; d[2] = (__bf16)v.z; d[3] = (__bf16)v.w;
  }
  __syncthreads();

  // ---- Phase A: p[l, n] = (1/H) sum_h exp(S_h - m)/s ----
  float pacc[2][8];
#pragma unroll
  for (int lt = 0; lt < 2; ++lt)
#pragma unroll
    for (int r = 0; r < 8; ++r) pacc[lt][r] = 0.0f;

  const int nrow = wave * 16 + mrow;  // this wave's 16-n slice; lane = column
  for (int h = 0; h < NH; ++h) {
    v16bf a0k0, a0k1, a1k0, a1k1, bk0, bk1;
    {
      V16 t;
      const __bf16* q0 = qs + (size_t)mrow * QP + h * HD;
      const __bf16* q1 = qs + (size_t)(16 + mrow) * QP + h * HD;
      t.q[0] = *(const uint4v*)(q0 + kb);      t.q[1] = *(const uint4v*)(q0 + 16 + kb); a0k0 = t.v;
      t.q[0] = *(const uint4v*)(q0 + 32 + kb); t.q[1] = *(const uint4v*)(q0 + 48 + kb); a0k1 = t.v;
      t.q[0] = *(const uint4v*)(q1 + kb);      t.q[1] = *(const uint4v*)(q1 + 16 + kb); a1k0 = t.v;
      t.q[0] = *(const uint4v*)(q1 + 32 + kb); t.q[1] = *(const uint4v*)(q1 + 48 + kb); a1k1 = t.v;
      const __bf16* kr = xs + (size_t)nrow * XP + h * HD + kb;
      t.q[0] = *(const uint4v*)(kr);      t.q[1] = *(const uint4v*)(kr + 16); bk0 = t.v;
      t.q[0] = *(const uint4v*)(kr + 32); t.q[1] = *(const uint4v*)(kr + 48); bk1 = t.v;
    }
    v8f c0 = {}, c1 = {};
    c0 = wmma_bf16(a0k0, bk0, c0);
    c0 = wmma_bf16(a0k1, bk1, c0);
    c1 = wmma_bf16(a1k0, bk0, c1);
    c1 = wmma_bf16(a1k1, bk1, c1);
#pragma unroll
    for (int r = 0; r < 8; ++r) {
      float2 s0 = st[h * LAT + (r + addl)];
      float2 s1 = st[h * LAT + (16 + r + addl)];
      pacc[0][r] += __expf(c0[r] - s0.x) / s0.y;
      pacc[1][r] += __expf(c1[r] - s1.x) / s1.y;
    }
  }
  {
    const float inv = 1.0f / NH;
    int ncol = wave * 16 + mrow;
#pragma unroll
    for (int lt = 0; lt < 2; ++lt)
#pragma unroll
      for (int r = 0; r < 8; ++r)
        ps[(size_t)(lt * 16 + r + addl) * PP + ncol] = pacc[lt][r] * inv;
  }
  __syncthreads();

  // ---- Phase B: y[l, c] += p[l, 0:128] @ x_tile[0:128, c] ----
  v16bf pa[2][4];
#pragma unroll
  for (int lt = 0; lt < 2; ++lt) {
    const float* pr = ps + (size_t)(lt * 16 + mrow) * PP;
#pragma unroll
    for (int kc = 0; kc < 4; ++kc) {
#pragma unroll
      for (int i = 0; i < 16; ++i) {
        int k = kc * 32 + ((i < 8) ? (kb + i) : (16 + kb + i - 8));
        pa[lt][kc][i] = (__bf16)pr[k];
      }
    }
  }

  float* yb = y + (size_t)b * LAT * DIM;
  for (int ct = wave; ct < DIM / 16; ct += 8) {   // 6 c-tiles per wave
    int ncol = ct * 16 + mrow;
    v8f c0 = {}, c1 = {};
#pragma unroll
    for (int kc = 0; kc < 4; ++kc) {
      v16bf bx;
      const __bf16* col = xs + (size_t)(kc * 32 + kb) * XP + ncol;
#pragma unroll
      for (int i = 0; i < 16; ++i) bx[i] = col[(size_t)((i < 8) ? i : i + 8) * XP];
      c0 = wmma_bf16(pa[0][kc], bx, c0);
      c1 = wmma_bf16(pa[1][kc], bx, c1);
    }
#pragma unroll
    for (int r = 0; r < 8; ++r) {
      atomicAdd(&yb[(size_t)(r + addl) * DIM + ncol], c0[r]);        // l = r+addl
      atomicAdd(&yb[(size_t)(16 + r + addl) * DIM + ncol], c1[r]);   // l = 16+r+addl
    }
  }
}

// ---------------------------------------------------------------------------
// K3: out[b, l*24+j] = y[b,l,:] . v_w[l*24+j,:] + v_b[l*24+j]   (tiny)
// ---------------------------------------------------------------------------
__global__ __launch_bounds__(32) void k3_out(
    const float* __restrict__ y, const float* __restrict__ v_w,
    const float* __restrict__ v_b, float* __restrict__ out) {
  const int b = blockIdx.x, l = blockIdx.y;
  const int j = threadIdx.x;
  if (j >= CHUNK) return;
  const int row = l * CHUNK + j;
  const float* yy = y + ((size_t)b * LAT + l) * DIM;
  const float* w  = v_w + (size_t)row * DIM;
  float acc = v_b[row];
  for (int c = 0; c < DIM; c += 4) {
    float4 a  = *(const float4*)(yy + c);
    float4 wv = *(const float4*)(w + c);
    acc += a.x * wv.x + a.y * wv.y + a.z * wv.z + a.w * wv.w;
  }
  out[(size_t)b * DIM + row] = acc;
}

extern "C" void kernel_launch(void* const* d_in, const int* in_sizes, int n_in,
                              void* d_out, int out_size, void* d_ws, size_t ws_size,
                              hipStream_t stream) {
  (void)in_sizes; (void)n_in; (void)out_size; (void)ws_size;
  const float* x      = (const float*)d_in[0];
  const float* latent = (const float*)d_in[1];
  const float* v_w    = (const float*)d_in[2];
  const float* v_b    = (const float*)d_in[3];
  float* out = (float*)d_out;

  float2* stats = (float2*)d_ws;                                   // 96 KB
  float*  y = (float*)((char*)d_ws + (size_t)NB * NH * LAT * sizeof(float2)); // 3 MB

  hipMemsetAsync(y, 0, (size_t)NB * LAT * DIM * sizeof(float), stream);

  k1_attn_stats<<<dim3(NB, NH), 256, 0, stream>>>(x, latent, stats);

  size_t sm = (size_t)128 * XP * 2 + (size_t)LAT * QP * 2 +
              (size_t)LAT * PP * 4 + (size_t)NH * LAT * sizeof(float2); // 268,288 B
  k2_probs_y<<<dim3(NB, NSEQ / 128), 256, sm, stream>>>(x, latent, stats, y);

  k3_out<<<dim3(NB, LAT), 32, 0, stream>>>(y, v_w, v_b, out);
}
